// MultiHeadAttention_50199577755877
// MI455X (gfx1250) — matrix operand order repaired and checked by reading, not compiled, not run on hardware
//
#include <hip/hip_runtime.h>
#include <hip/hip_bf16.h>
#include <math.h>

typedef __attribute__((ext_vector_type(16))) _Float16 v16h;
typedef __attribute__((ext_vector_type(8)))  float    v8f;

union Frag16 { v16h h; unsigned int u[8]; };
union Pack2  { unsigned int u; _Float16 h[2]; };

#define S_LEN   2048
#define D_MODEL 1024
#define NHEAD   16
#define HDIM    64
#define BATCH   2
#define MTOT    (BATCH * S_LEN)               // 4096
#define OUT_ELEMS ((size_t)MTOT * D_MODEL)    // out part of d_out

#define TILE_M 128
#define TILE_N 128
#define TILE_K 32
#define LDSS   40   // f16 row stride (+8 pad => 80B, 16B-aligned, bank-conflict free)
#define KSTRIDE 72  // f16 row stride for 64-wide K rows (144B, conflict-free)

// ---- CDNA5 async global->LDS copy (ASYNCcnt-tracked, GVS addressing) -------
__device__ __forceinline__ void async_copy_b128(unsigned lds_off, unsigned goff,
                                                const void* sbase)
{
    asm volatile("global_load_async_to_lds_b128 %0, %1, %2"
                 :: "v"(lds_off), "v"(goff), "s"(sbase)
                 : "memory");
}
__device__ __forceinline__ void wait_async0()
{
    asm volatile("s_wait_asynccnt 0x0" ::: "memory");
}
__device__ __forceinline__ unsigned lds_addr(const void* p)
{
    return (unsigned)(size_t)p;   // LDS aperture: addr[31:0] == LDS byte offset
}
// pin instruction regions: keep WMMAs ahead of the async wait
#define SCHED_FENCE() __builtin_amdgcn_sched_barrier(0)

// --------------------------------------------------------------------------
// one-time f32 -> f16 conversion (flat), 8 elems/thread
// --------------------------------------------------------------------------
__global__ __launch_bounds__(256)
void cvt_kernel(const float* __restrict__ src, _Float16* __restrict__ dst)
{
    const size_t i = ((size_t)blockIdx.x * 256 + threadIdx.x) * 8;
    const float4 a = *(const float4*)(src + i);
    const float4 b = *(const float4*)(src + i + 4);
    union { uint4 v; _Float16 h[8]; } o;
    o.h[0] = (_Float16)a.x; o.h[1] = (_Float16)a.y;
    o.h[2] = (_Float16)a.z; o.h[3] = (_Float16)a.w;
    o.h[4] = (_Float16)b.x; o.h[5] = (_Float16)b.y;
    o.h[6] = (_Float16)b.z; o.h[7] = (_Float16)b.w;
    *(uint4*)(dst + i) = o.v;
}

// --------------------------------------------------------------------------
// one-time W (KxN f32) -> W^T (NxK f16), 32x32 LDS tile
// --------------------------------------------------------------------------
__global__ __launch_bounds__(256)
void cvtT_kernel(const float* __restrict__ W, _Float16* __restrict__ WT,
                 int K, int N)
{
    __shared__ _Float16 t[32][36];
    const int k0 = blockIdx.y * 32, n0 = blockIdx.x * 32;
    const int r  = threadIdx.x >> 3;
    const int c4 = (threadIdx.x & 7) * 4;
    const float4 f = *(const float4*)(W + (size_t)(k0 + r) * N + n0 + c4);
    t[r][c4 + 0] = (_Float16)f.x; t[r][c4 + 1] = (_Float16)f.y;
    t[r][c4 + 2] = (_Float16)f.z; t[r][c4 + 3] = (_Float16)f.w;
    __syncthreads();
    union { uint2 v; _Float16 h[4]; } o;
    o.h[0] = t[c4 + 0][r]; o.h[1] = t[c4 + 1][r];
    o.h[2] = t[c4 + 2][r]; o.h[3] = t[c4 + 3][r];
    *(uint2*)(WT + (size_t)(n0 + r) * K + k0 + c4) = o.v;
}

// --------------------------------------------------------------------------
// GEMM: Y = (A @ W + bias) * scale,  A: MxK f16, WT: NxK f16 (W transposed)
//   MODE 0: f16 head-split (B,H,S,Hd)             (Q,K projections)
//   MODE 1: f32 flat MxN                          (output projection)
//   MODE 2: f16 transposed head-split (B,H,Hd,S)  (V projection)
// Double-buffered async global->LDS staging.
// --------------------------------------------------------------------------
template<int MODE>
__global__ __launch_bounds__(256)
void mha_gemm_f16(const _Float16* __restrict__ A, const _Float16* __restrict__ WT,
                  const float* __restrict__ bias,
                  _Float16* __restrict__ out_h, float* __restrict__ out_f,
                  int M, int N, int K, float scale)
{
    __shared__ _Float16 lA[2][TILE_M * LDSS];
    __shared__ _Float16 lB[2][TILE_N * LDSS];

    const int tid  = threadIdx.x;
    const int wid  = tid >> 5;
    const int lane = tid & 31;
    const int lm   = lane & 15;
    const int hf   = lane >> 4;
    const int wm   = wid & 3;
    const int wn   = wid >> 2;
    const int m0   = blockIdx.x * TILE_M;
    const int n0   = blockIdx.y * TILE_N;

    auto stage = [&](int buf, int k0) {
        const int row = tid >> 1;            // 0..127
        const int kq  = (tid & 1) << 4;      // 0 or 16 (f16 units)
        const unsigned la = lds_addr(&lA[buf][row * LDSS + kq]);
        const unsigned ga = (unsigned)(((unsigned)(m0 + row) * K + k0 + kq) * 2);
        async_copy_b128(la,      ga,      A);
        async_copy_b128(la + 16, ga + 16, A);
        const unsigned lb = lds_addr(&lB[buf][row * LDSS + kq]);
        const unsigned gb = (unsigned)(((unsigned)(n0 + row) * K + k0 + kq) * 2);
        async_copy_b128(lb,      gb,      WT);
        async_copy_b128(lb + 16, gb + 16, WT);
    };

    v8f acc[2][4];
    #pragma unroll
    for (int i = 0; i < 2; ++i)
        #pragma unroll
        for (int j = 0; j < 4; ++j) acc[i][j] = (v8f)0.0f;

    int cur = 0;
    stage(0, 0);
    wait_async0();
    __syncthreads();

    for (int k0 = 0; k0 < K; k0 += TILE_K) {
        if (k0 + TILE_K < K) stage(cur ^ 1, k0 + TILE_K);   // prefetch next tile

        Frag16 afr[2], bfr[4];
        #pragma unroll
        for (int mt = 0; mt < 2; ++mt) {
            const int row = wm * 32 + mt * 16 + lm;
            const unsigned int* src = (const unsigned int*)&lA[cur][row * LDSS];
            #pragma unroll
            for (int j = 0; j < 8; ++j) {
                const int kk = hf * 16 + 2 * j; // contiguous k = 16*hf + s (same as B)
                afr[mt].u[j] = src[kk >> 1];
            }
        }
        #pragma unroll
        for (int nt = 0; nt < 4; ++nt) {
            const int col = wn * 64 + nt * 16 + lm;
            const unsigned int* src = (const unsigned int*)&lB[cur][col * LDSS];
            #pragma unroll
            for (int v = 0; v < 8; ++v)                                  // contiguous k = 16*hf + s (same as A)
                bfr[nt].u[v] = src[hf * 8 + v];
        }
        #pragma unroll
        for (int mt = 0; mt < 2; ++mt)
            #pragma unroll
            for (int nt = 0; nt < 4; ++nt)
                acc[mt][nt] = __builtin_amdgcn_wmma_f32_16x16x32_f16(
                    false, afr[mt].h, false, bfr[nt].h,
                    (short)0, acc[mt][nt], false, false);

        SCHED_FENCE();      // keep WMMAs ahead of the async wait (overlap copy)
        wait_async0();      // next buffer fully staged
        __syncthreads();
        cur ^= 1;
    }

    #pragma unroll
    for (int mt = 0; mt < 2; ++mt) {
        #pragma unroll
        for (int nt = 0; nt < 4; ++nt) {
            const int n  = n0 + wn * 64 + nt * 16 + lm;
            const float bv = bias[n];
            #pragma unroll
            for (int r = 0; r < 8; ++r) {
                const int m = m0 + wm * 32 + mt * 16 + hf * 8 + r;
                const float val = (acc[mt][nt][r] + bv) * scale;
                const int b = m >> 11, s = m & (S_LEN - 1);
                const int h = n >> 6,  d = n & (HDIM - 1);
                if (MODE == 0) {
                    out_h[((((size_t)b * NHEAD + h) * S_LEN + s) << 6) + d] =
                        (_Float16)val;
                } else if (MODE == 1) {
                    out_f[(size_t)m * N + n] = val;
                } else { // MODE 2: transposed head-split (B,H,Hd,S)
                    out_h[(((size_t)b * NHEAD + h) * HDIM + d) * S_LEN + s] =
                        (_Float16)val;
                }
            }
        }
    }
}

// --------------------------------------------------------------------------
// Fused scores + softmax: one (b,h, 16-row q-tile) per block.
// Whole 16x2048 score row-set lives in registers (8 waves x 16 accs).
// K rows streamed in 128-row chunks via double-buffered async-to-LDS.
// Writes NORMALIZED attn f32 exactly once.
// grid: (S/16, B*H), block 256
// --------------------------------------------------------------------------
__global__ __launch_bounds__(256)
void mha_scores_softmax_kernel(const _Float16* __restrict__ qh,
                               const _Float16* __restrict__ kh,
                               float* __restrict__ attn)
{
    __shared__ _Float16 lK[2][128 * KSTRIDE];  // [k-row][d], 18KB each
    __shared__ float red[8][16];
    __shared__ float rowstat[16];

    const int tid  = threadIdx.x;
    const int wid  = tid >> 5;
    const int lane = tid & 31;
    const int lm   = lane & 15;
    const int hf   = lane >> 4;
    const int qt   = blockIdx.x;
    const int bh   = blockIdx.y;

    const _Float16* Q  = qh + (size_t)bh * S_LEN * HDIM;
    const _Float16* Kp = kh + (size_t)bh * S_LEN * HDIM;

    // Q fragments for this q-tile (K-dim 64 -> 2 fragments), loaded once
    Frag16 a[2];
    #pragma unroll
    for (int i = 0; i < 2; ++i) {
        const unsigned int* qs =
            (const unsigned int*)(Q + (size_t)(qt * 16 + lm) * HDIM + i * 32);
        #pragma unroll
        for (int j = 0; j < 8; ++j) {
            const int kk = hf * 16 + 2 * j;
            a[i].u[j] = qs[kk >> 1];
        }
    }

    auto stage = [&](int buf, int c) {
        const int row = tid >> 1;            // 0..127
        const int dq  = (tid & 1) << 5;      // 0 or 32 (f16 units)
        const unsigned la = lds_addr(&lK[buf][row * KSTRIDE + dq]);
        const unsigned ga = (unsigned)(((unsigned)(c * 128 + row) * HDIM + dq) * 2);
        async_copy_b128(la,      ga,      Kp);
        async_copy_b128(la + 16, ga + 16, Kp);
        async_copy_b128(la + 32, ga + 32, Kp);
        async_copy_b128(la + 48, ga + 48, Kp);
    };

    v8f acc[16];
    #pragma unroll
    for (int c = 0; c < 16; ++c) acc[c] = (v8f)0.0f;

    stage(0, 0);
    wait_async0();
    __syncthreads();

    #pragma unroll
    for (int c = 0; c < 16; ++c) {
        if (c < 15) stage((c + 1) & 1, c + 1);
        const int krow = wid * 16 + lm;      // this wave's 16-col slice
        #pragma unroll
        for (int i = 0; i < 2; ++i) {
            Frag16 b;
            const unsigned int* bs = (const unsigned int*)
                &lK[c & 1][krow * KSTRIDE + i * 32 + hf * 16];
            #pragma unroll
            for (int v = 0; v < 8; ++v) b.u[v] = bs[v];
            acc[c] = __builtin_amdgcn_wmma_f32_16x16x32_f16(
                false, a[i].h, false, b.h, (short)0, acc[c], false, false);
        }
        SCHED_FENCE();      // WMMAs stay ahead of the wait -> copy/compute overlap
        wait_async0();
        __syncthreads();
    }

    // ---- softmax over the 16 rows (each lane holds 8 rows x 16 acc slices)
    float pm[8];
    #pragma unroll
    for (int p = 0; p < 8; ++p) {
        pm[p] = acc[0][p];
        #pragma unroll
        for (int c = 1; c < 16; ++c) pm[p] = fmaxf(pm[p], acc[c][p]);
    }
    #pragma unroll
    for (int off = 1; off <= 8; off <<= 1)
        #pragma unroll
        for (int p = 0; p < 8; ++p)
            pm[p] = fmaxf(pm[p], __shfl_xor(pm[p], off, 32));
    if (lm == 0) {
        #pragma unroll
        for (int p = 0; p < 8; ++p) red[wid][p + 8 * hf] = pm[p];
    }
    __syncthreads();
    if (tid < 16) {
        float m = red[0][tid];
        #pragma unroll
        for (int w = 1; w < 8; ++w) m = fmaxf(m, red[w][tid]);
        rowstat[tid] = m;
    }
    __syncthreads();

    float ps[8];
    #pragma unroll
    for (int p = 0; p < 8; ++p) {
        const float rm = rowstat[p + 8 * hf];
        ps[p] = 0.0f;
        #pragma unroll
        for (int c = 0; c < 16; ++c) {
            acc[c][p] = __expf(acc[c][p] - rm);
            ps[p] += acc[c][p];
        }
    }
    #pragma unroll
    for (int off = 1; off <= 8; off <<= 1)
        #pragma unroll
        for (int p = 0; p < 8; ++p)
            ps[p] += __shfl_xor(ps[p], off, 32);
    if (lm == 0) {
        #pragma unroll
        for (int p = 0; p < 8; ++p) red[wid][p + 8 * hf] = ps[p];
    }
    __syncthreads();
    if (tid < 16) {
        float s = 0.0f;
        #pragma unroll
        for (int w = 0; w < 8; ++w) s += red[w][tid];
        rowstat[tid] = 1.0f / s;
    }
    __syncthreads();

    // ---- write normalized attn
    #pragma unroll
    for (int p = 0; p < 8; ++p) {
        const float inv = rowstat[p + 8 * hf];
        float* o = attn + ((size_t)bh * S_LEN + qt * 16 + p + 8 * hf) * S_LEN
                        + wid * 16 + lm;
        #pragma unroll
        for (int c = 0; c < 16; ++c)
            o[c * 128] = acc[c][p] * inv;
    }
}

// --------------------------------------------------------------------------
// ctx = attn @ V; V held transposed head-split (B,H,Hd,S) so the V tile is a
// contiguous row copy -> async-to-LDS, double buffered. Writes f16 (B,S,D).
// grid: (S/128, B*H)
// --------------------------------------------------------------------------
__global__ __launch_bounds__(256)
void mha_ctx_kernel(const float* __restrict__ attn,
                    const _Float16* __restrict__ vt,
                    _Float16* __restrict__ ctx)
{
    __shared__ _Float16 lV[2][HDIM * LDSS];   // [d][k]

    const int tid  = threadIdx.x;
    const int wid  = tid >> 5;
    const int lane = tid & 31;
    const int lm   = lane & 15;
    const int hf   = lane >> 4;
    const int bh   = blockIdx.y;
    const int qbase = blockIdx.x * 128 + wid * 16;

    const float*    Ap = attn + (size_t)bh * S_LEN * S_LEN;
    const _Float16* Vb = vt   + (size_t)bh * HDIM * S_LEN;

    auto stage = [&](int buf, int k0) {
        const int row = tid >> 2;           // 0..63 (d)
        const int kq  = (tid & 3) << 3;     // 0,8,16,24 (f16 units)
        const unsigned la = lds_addr(&lV[buf][row * LDSS + kq]);
        const unsigned ga = (unsigned)(((unsigned)row * S_LEN + k0 + kq) * 2);
        async_copy_b128(la, ga, Vb);
    };

    v8f acc[4];
    #pragma unroll
    for (int i = 0; i < 4; ++i) acc[i] = (v8f)0.0f;

    int cur = 0;
    stage(0, 0);
    wait_async0();
    __syncthreads();

    for (int k0 = 0; k0 < S_LEN; k0 += 32) {
        if (k0 + 32 < S_LEN) stage(cur ^ 1, k0 + 32);

        Frag16 a;
        const float* as = Ap + (size_t)(qbase + lm) * S_LEN + k0;
        #pragma unroll
        for (int j = 0; j < 8; ++j) {
            const int kk = hf * 16 + 2 * j;
            const float2 f = *(const float2*)(as + kk);
            Pack2 pp; pp.h[0] = (_Float16)f.x; pp.h[1] = (_Float16)f.y;
            a.u[j] = pp.u;
        }
        #pragma unroll
        for (int nt = 0; nt < 4; ++nt) {
            Frag16 b;
            const unsigned int* bs =
                (const unsigned int*)&lV[cur][(nt * 16 + lm) * LDSS];
            #pragma unroll
            for (int v = 0; v < 8; ++v) b.u[v] = bs[hf * 8 + v];
            acc[nt] = __builtin_amdgcn_wmma_f32_16x16x32_f16(
                false, a.h, false, b.h, (short)0, acc[nt], false, false);
        }

        SCHED_FENCE();      // WMMAs stay ahead of the wait -> copy/compute overlap
        wait_async0();
        __syncthreads();
        cur ^= 1;
    }

    const int b_ = bh >> 4, h_ = bh & (NHEAD - 1);
    #pragma unroll
    for (int nt = 0; nt < 4; ++nt)
        #pragma unroll
        for (int r = 0; r < 8; ++r) {
            const int s = qbase + hf * 8 + r;
            ctx[((size_t)b_ * S_LEN + s) * D_MODEL + h_ * HDIM + nt * 16 + lm] =
                (_Float16)acc[nt][r];
        }
}

// --------------------------------------------------------------------------
extern "C" void kernel_launch(void* const* d_in, const int* in_sizes, int n_in,
                              void* d_out, int out_size, void* d_ws, size_t ws_size,
                              hipStream_t stream)
{
    const float* q  = (const float*)d_in[0];
    const float* k  = (const float*)d_in[1];
    const float* v  = (const float*)d_in[2];
    const float* wq = (const float*)d_in[3];
    const float* bq = (const float*)d_in[4];
    const float* wk = (const float*)d_in[5];
    const float* bk = (const float*)d_in[6];
    const float* wv = (const float*)d_in[7];
    const float* bv = (const float*)d_in[8];
    const float* wo = (const float*)d_in[9];
    const float* bo = (const float*)d_in[10];

    float* out  = (float*)d_out;                 // (B,S,D)
    float* attn = (float*)d_out + OUT_ELEMS;     // (B,H,S,S)

    // workspace layout (bytes)
    const size_t MB = 1024 * 1024;
    char* ws = (char*)d_ws;
    _Float16* qf   = (_Float16*)(ws);            // 8MB  f16 flat inputs
    _Float16* kf   = (_Float16*)(ws + 8  * MB);
    _Float16* vf   = (_Float16*)(ws + 16 * MB);
    _Float16* wqT  = (_Float16*)(ws + 24 * MB);  // 2MB each, W^T f16
    _Float16* wkT  = (_Float16*)(ws + 26 * MB);
    _Float16* wvT  = (_Float16*)(ws + 28 * MB);
    _Float16* woT  = (_Float16*)(ws + 30 * MB);
    _Float16* qh   = (_Float16*)(ws + 32 * MB);  // head-split (B,H,S,Hd)
    _Float16* kh   = (_Float16*)(ws + 40 * MB);
    _Float16* vt   = (_Float16*)(ws + 48 * MB);  // transposed head-split (B,H,Hd,S)
    _Float16* ctxh = (_Float16*)(ws + 56 * MB);  // (B,S,D) f16

    const unsigned cvtBlocks = (unsigned)(OUT_ELEMS / (256 * 8));   // 2048
    cvt_kernel<<<cvtBlocks, 256, 0, stream>>>(q, qf);
    cvt_kernel<<<cvtBlocks, 256, 0, stream>>>(k, kf);
    cvt_kernel<<<cvtBlocks, 256, 0, stream>>>(v, vf);

    const dim3 gT(D_MODEL / 32, D_MODEL / 32);
    cvtT_kernel<<<gT, 256, 0, stream>>>(wq, wqT, D_MODEL, D_MODEL);
    cvtT_kernel<<<gT, 256, 0, stream>>>(wk, wkT, D_MODEL, D_MODEL);
    cvtT_kernel<<<gT, 256, 0, stream>>>(wv, wvT, D_MODEL, D_MODEL);
    cvtT_kernel<<<gT, 256, 0, stream>>>(wo, woT, D_MODEL, D_MODEL);

    const dim3 gGemm(MTOT / TILE_M, D_MODEL / TILE_N);   // 32 x 8
    const float qscale = 1.0f / sqrtf((float)HDIM);

    mha_gemm_f16<0><<<gGemm, 256, 0, stream>>>(qf, wqT, bq, qh, nullptr,
                                               MTOT, D_MODEL, D_MODEL, qscale);
    mha_gemm_f16<0><<<gGemm, 256, 0, stream>>>(kf, wkT, bk, kh, nullptr,
                                               MTOT, D_MODEL, D_MODEL, 1.0f);
    mha_gemm_f16<2><<<gGemm, 256, 0, stream>>>(vf, wvT, bv, vt, nullptr,
                                               MTOT, D_MODEL, D_MODEL, 1.0f);

    // fused scores + softmax: writes normalized attn once
    mha_scores_softmax_kernel<<<dim3(S_LEN / 16, BATCH * NHEAD),
                                256, 0, stream>>>(qh, kh, attn);

    mha_ctx_kernel<<<dim3(S_LEN / 128, BATCH * NHEAD), 256, 0, stream>>>(attn, vt, ctxh);

    mha_gemm_f16<1><<<gGemm, 256, 0, stream>>>(ctxh, woT, bo, nullptr, out,
                                               MTOT, D_MODEL, D_MODEL, 1.0f);
}
